// RepeatedAttention_70660801954634
// MI455X (gfx1250) — matrix operand order repaired and checked by reading, not compiled, not run on hardware
//
#include <hip/hip_runtime.h>
#include <hip/hip_bf16.h>
#include <math.h>
#include <stdint.h>

typedef __attribute__((ext_vector_type(16))) _Float16 v16h;
typedef __attribute__((ext_vector_type(8)))  _Float16 v8h;
typedef __attribute__((ext_vector_type(8)))  float    v8f;

#define S_LEN 2048
#define HID   2048
#define NH    16
#define NKV   4
#define HD    128
#define WINSZ 512
#define QK_SCALE 0.08838834764831845f  // 1/sqrt(128)

// ---------------------------------------------------------------------------
// CDNA5 async global->LDS copy (ASYNCcnt-tracked). lds_addr = LDS byte offset
// (low 32 bits of a generic pointer to __shared__), gaddr = 64-bit global.
__device__ __forceinline__ void async_ld_b128(uint32_t lds_addr, const void* gaddr) {
  asm volatile("global_load_async_to_lds_b128 %0, %1, off"
               :: "v"(lds_addr), "v"(gaddr)
               : "memory");
}
#define WAIT_ASYNC(n) asm volatile("s_wait_asynccnt %0" :: "n"(n) : "memory")

__device__ __forceinline__ uint32_t lds_off(const void* p) {
  return (uint32_t)(uintptr_t)p;
}

// ---------------------------------------------------------------------------
// elementwise f32 -> f16 copy
__global__ __launch_bounds__(256) void cvt_f16_kernel(const float* __restrict__ src,
                                                      _Float16* __restrict__ dst, int n) {
  int i = blockIdx.x * 256 + threadIdx.x;
  if (i < n) dst[i] = (_Float16)src[i];
}

// transpose + convert: src is R x C (f32, row-major) -> dst is C x R (f16)
__global__ __launch_bounds__(256) void transpose_cvt_kernel(const float* __restrict__ src,
                                                            _Float16* __restrict__ dst,
                                                            int R, int C) {
  __shared__ float tile[32][33];
  int c0 = blockIdx.x * 32, r0 = blockIdx.y * 32;
  int tx = threadIdx.x, ty = threadIdx.y;
#pragma unroll
  for (int i = 0; i < 4; ++i)
    tile[ty + i * 8][tx] = src[(size_t)(r0 + ty + i * 8) * C + c0 + tx];
  __syncthreads();
#pragma unroll
  for (int i = 0; i < 4; ++i)
    dst[(size_t)(c0 + ty + i * 8) * R + r0 + tx] = (_Float16)tile[tx][ty + i * 8];
}

// transpose f16 -> f16: src R x C -> dst C x R
__global__ __launch_bounds__(256) void transpose_f16_kernel(const _Float16* __restrict__ src,
                                                            _Float16* __restrict__ dst,
                                                            int R, int C) {
  __shared__ _Float16 tile[32][33];
  int c0 = blockIdx.x * 32, r0 = blockIdx.y * 32;
  int tx = threadIdx.x, ty = threadIdx.y;
#pragma unroll
  for (int i = 0; i < 4; ++i)
    tile[ty + i * 8][tx] = src[(size_t)(r0 + ty + i * 8) * C + c0 + tx];
  __syncthreads();
#pragma unroll
  for (int i = 0; i < 4; ++i)
    dst[(size_t)(c0 + ty + i * 8) * R + r0 + tx] = tile[tx][ty + i * 8];
}

// ---------------------------------------------------------------------------
// GEMM: C[M,N] = A[M,K](f16) * Bt[N,K](f16)^T + bias[N]
// block tile 128x128, BK=32, 8 waves in 4x2 grid, each wave 32x64 (2x4 wmma).
// Double-buffered LDS, tiles staged with async global->LDS; peeled pipeline.
__global__ __launch_bounds__(256) void gemm_kernel(const _Float16* __restrict__ A,
                                                   const _Float16* __restrict__ Bt,
                                                   const float* __restrict__ bias,
                                                   void* __restrict__ Cout,
                                                   int M, int N, int K, int out_f16) {
  __shared__ alignas(32) _Float16 Al[2][128 * 32];
  __shared__ alignas(32) _Float16 Bl[2][128 * 32];
  const int tid  = threadIdx.x;
  const int wave = tid >> 5, lane = tid & 31;
  const int lg = lane >> 4, ln = lane & 15;
  const int wm = wave & 3, wn = wave >> 2;  // 4x2 wave grid
  const size_t mBase = (size_t)blockIdx.x * 128;
  const size_t nBase = (size_t)blockIdx.y * 128;

  // per-thread staging slots: A: 16 halfs (2 x b128), B: 16 halfs (2 x b128)
  const int srow = tid >> 1, sc = (tid & 1) * 16;
  const _Float16* Ag = A + (mBase + srow) * (size_t)K + sc;
  const _Float16* Bg = Bt + (nBase + srow) * (size_t)K + sc;

  auto stage = [&](int buf, int k0) {
    uint32_t la = lds_off(&Al[buf][srow * 32 + sc]);
    async_ld_b128(la,      Ag + k0);
    async_ld_b128(la + 16, Ag + k0 + 8);
    uint32_t lb = lds_off(&Bl[buf][srow * 32 + sc]);
    async_ld_b128(lb,      Bg + k0);
    async_ld_b128(lb + 16, Bg + k0 + 8);
  };

  v8f acc[2][4];
#pragma unroll
  for (int i = 0; i < 2; ++i)
#pragma unroll
    for (int j = 0; j < 4; ++j) acc[i][j] = v8f{};

  auto compute = [&](int cur) {
    union Frag { v16h v; v8h h8[2]; };
    Frag aF[2];
    v16h bF[4];
#pragma unroll
    for (int sm = 0; sm < 2; ++sm) {
      const _Float16* ar = &Al[cur][(wm * 32 + sm * 16 + ln) * 32];
      aF[sm].h8[0] = *(const v8h*)(ar + lg * 16);       // k = 16g + 0..7
      aF[sm].h8[1] = *(const v8h*)(ar + lg * 16 + 8);   // k = 16g + 8..15
    }
#pragma unroll
    for (int sn = 0; sn < 4; ++sn) {
      const _Float16* br = &Bl[cur][(wn * 64 + sn * 16 + ln) * 32];
      bF[sn] = *(const v16h*)(br + lg * 16);            // k = 16g + 0..15
    }
#pragma unroll
    for (int sm = 0; sm < 2; ++sm)
#pragma unroll
      for (int sn = 0; sn < 4; ++sn)
        acc[sm][sn] = __builtin_amdgcn_wmma_f32_16x16x32_f16(
            false, aF[sm].v, false, bF[sn], (short)0, acc[sm][sn], false, false);
  };

  const int nk = K >> 5;  // >= 2 for all our shapes
  stage(0, 0);
  for (int i = 0; i < nk - 1; ++i) {
    stage((i + 1) & 1, (i + 1) << 5);
    WAIT_ASYNC(4);        // previous batch complete, next still in flight
    __syncthreads();
    compute(i & 1);
    __syncthreads();
  }
  WAIT_ASYNC(0);
  __syncthreads();
  compute((nk - 1) & 1);

#pragma unroll
  for (int sm = 0; sm < 2; ++sm)
#pragma unroll
    for (int sn = 0; sn < 4; ++sn) {
      int n = (int)nBase + wn * 64 + sn * 16 + ln;
      float bv = bias ? bias[n] : 0.f;
#pragma unroll
      for (int r = 0; r < 8; ++r) {
        size_t m = mBase + wm * 32 + sm * 16 + r + 8 * lg;
        float v = acc[sm][sn][r] + bv;
        if (out_f16) ((_Float16*)Cout)[m * N + n] = (_Float16)v;
        else         ((float*)Cout)[m * N + n] = v;
      }
    }
}

// ---------------------------------------------------------------------------
// RoPE for Q: Qp[S,4096] f16 -> Qh[2][H][S][D] f16
__global__ __launch_bounds__(128) void rope_q_kernel(const _Float16* __restrict__ Qp,
                                                     const float* __restrict__ cosb,
                                                     const float* __restrict__ sinb,
                                                     _Float16* __restrict__ Qh) {
  int s = blockIdx.x, vh = blockIdx.y, d = threadIdx.x;
  int t = vh >> 4, h = vh & 15;
  const _Float16* row = Qp + (size_t)s * (2 * NH * HD) + t * (NH * HD) + h * HD;
  float x  = (float)row[d];
  float xr = (d < 64) ? -(float)row[d + 64] : (float)row[d - 64];
  float y  = x * cosb[s * HD + d] + xr * sinb[s * HD + d];
  Qh[(((size_t)(t * NH + h)) * S_LEN + s) * HD + d] = (_Float16)y;
}

// RoPE for K: Kp[S,512] f16 -> Kh[KV][S][D] f16
__global__ __launch_bounds__(128) void rope_k_kernel(const _Float16* __restrict__ Kp,
                                                     const float* __restrict__ cosb,
                                                     const float* __restrict__ sinb,
                                                     _Float16* __restrict__ Kh) {
  int s = blockIdx.x, kv = blockIdx.y, d = threadIdx.x;
  const _Float16* row = Kp + (size_t)s * (NKV * HD) + kv * HD;
  float x  = (float)row[d];
  float xr = (d < 64) ? -(float)row[d + 64] : (float)row[d - 64];
  float y  = x * cosb[s * HD + d] + xr * sinb[s * HD + d];
  Kh[((size_t)kv * S_LEN + s) * HD + d] = (_Float16)y;
}

// ---------------------------------------------------------------------------
// Flash attention, dual mask (z=0: causal, z=1: causal+512 window, *tanh(gamma))
// block = 4 waves; wave w handles query rows [q0+16w, q0+16w+16).
// Scores computed as S^T = K_tile x Q^T so P^T(C-layout) repacks into the
// PV A-operand with lane-local f16 packing only. K/V tiles double-buffered
// in LDS via async global->LDS copies; peeled pipeline.
__global__ __launch_bounds__(128) void attn_kernel(const _Float16* __restrict__ Qh,
                                                   const _Float16* __restrict__ Kh,
                                                   const _Float16* __restrict__ Vt,
                                                   const float* __restrict__ gamma,
                                                   _Float16* __restrict__ Oc) {
  __shared__ alignas(32) _Float16 Kl[2][32 * 128];  // 32 keys x 128 d
  __shared__ alignas(32) _Float16 Vl[2][128 * 32];  // 128 d x 32 keys

  const int tid  = threadIdx.x;
  const int wave = tid >> 5, lane = tid & 31;
  const int lg = lane >> 4, ln = lane & 15;

  const int qt = blockIdx.x;   // query tile (64 rows)
  const int h  = blockIdx.y;   // head
  const int tt = blockIdx.z;   // 0 = causal, 1 = windowed
  const int kv = h >> 2;       // GQA group

  const int q0 = qt * 64;
  const int qw = q0 + wave * 16;
  const int q  = qw + ln;      // this lane's query index (stats)

  // Q fragments as B-operand: lane = query column, halves d = 32c + 16g + 0..15
  const _Float16* qrow = Qh + (((size_t)(tt * NH + h)) * S_LEN + q) * HD;
  v16h qf[4];
#pragma unroll
  for (int c = 0; c < 4; ++c) qf[c] = *(const v16h*)(qrow + c * 32 + lg * 16);

  v8f acc[8];
#pragma unroll
  for (int i = 0; i < 8; ++i) acc[i] = v8f{};
  float mrun = -__builtin_inff();
  float lsum = 0.f;

  int klo = 0;
  if (tt == 1) { klo = q0 - WINSZ; if (klo < 0) klo = 0; }
  const int khi = q0 + 64;
  const int nk  = (khi - klo) >> 5;   // >= 2 always

  const _Float16* Kbase = Kh + ((size_t)kv * S_LEN) * HD;
  const _Float16* Vbase = Vt + ((size_t)kv * HD) * S_LEN;

  // per-thread staging: K: one key row seg (64B = 4 x b128); V: one d row (64B)
  const int skey = tid >> 2, sdc = (tid & 3) * 32;
  auto stageKV = [&](int buf, int k0) {
    const _Float16* ks = Kbase + (size_t)(k0 + skey) * HD + sdc;
    uint32_t lk = lds_off(&Kl[buf][skey * 128 + sdc]);
    async_ld_b128(lk,      ks);
    async_ld_b128(lk + 16, ks + 8);
    async_ld_b128(lk + 32, ks + 16);
    async_ld_b128(lk + 48, ks + 24);
    const _Float16* vs = Vbase + (size_t)tid * S_LEN + k0;
    uint32_t lv = lds_off(&Vl[buf][tid * 32]);
    async_ld_b128(lv,      vs);
    async_ld_b128(lv + 16, vs + 8);
    async_ld_b128(lv + 32, vs + 16);
    async_ld_b128(lv + 48, vs + 24);
  };

  auto process = [&](int cur, int k0) {
    // S^T tiles: st[s2] covers local keys [16*s2, 16*s2+16)
    v8f st[2];
#pragma unroll
    for (int s2 = 0; s2 < 2; ++s2) {
      v8f a = v8f{};
#pragma unroll
      for (int c = 0; c < 4; ++c) {
        union { v16h v; v8h h8[2]; } ak;
        const _Float16* kr = &Kl[cur][(s2 * 16 + ln) * 128 + c * 32];
        ak.h8[0] = *(const v8h*)(kr + lg * 16);
        ak.h8[1] = *(const v8h*)(kr + lg * 16 + 8);
        a = __builtin_amdgcn_wmma_f32_16x16x32_f16(false, ak.v, false, qf[c],
                                                   (short)0, a, false, false);
      }
      st[s2] = a;
    }

    // scale + mask (lane = query q; vgpr r, group lg = key r+8g)
#pragma unroll
    for (int s2 = 0; s2 < 2; ++s2)
#pragma unroll
      for (int r = 0; r < 8; ++r) {
        int key = k0 + s2 * 16 + r + 8 * lg;
        float sv = st[s2][r] * QK_SCALE;
        bool ok = (key <= q) && (tt == 0 || key >= q - WINSZ);
        st[s2][r] = ok ? sv : -1.0e30f;
      }

    // online softmax
    float cm = st[0][0];
#pragma unroll
    for (int r = 0; r < 8; ++r) { cm = fmaxf(cm, st[0][r]); cm = fmaxf(cm, st[1][r]); }
    cm = fmaxf(cm, __shfl_xor(cm, 16, 32));
    float mnew = fmaxf(mrun, cm);
    float corr = __expf(mrun - mnew);
    mrun = mnew;

    union { v16h v; _Float16 e[16]; } pa;
    float rs = 0.f;
#pragma unroll
    for (int r = 0; r < 8; ++r) {
      float p0 = (st[0][r] <= -1.0e29f) ? 0.f : __expf(st[0][r] - mnew);
      float p1 = (st[1][r] <= -1.0e29f) ? 0.f : __expf(st[1][r] - mnew);
      pa.e[r] = (_Float16)p0;        // A halves 0..7  -> key 8g + r
      pa.e[8 + r] = (_Float16)p1;    // A halves 8..15 -> key 16 + 8g + r
      rs += p0 + p1;
    }
    rs += __shfl_xor(rs, 16, 32);
    lsum = lsum * corr + rs;

    // rescale O (acc vgpr r <-> query row r+8g; factor lives in lane r+8g)
#pragma unroll
    for (int r = 0; r < 8; ++r) {
      float fr = __shfl(corr, r + 8 * lg, 32);
#pragma unroll
      for (int nt = 0; nt < 8; ++nt) acc[nt][r] *= fr;
    }

    // O += P * V  (B-operand from Vt: lane = d column, halves = keys 8g+0..7, 16+8g+0..7)
#pragma unroll
    for (int nt = 0; nt < 8; ++nt) {
      union { v16h v; v8h h8[2]; } bv;
      bv.h8[0] = *(const v8h*)(&Vl[cur][(nt * 16 + ln) * 32 + lg * 8]);
      bv.h8[1] = *(const v8h*)(&Vl[cur][(nt * 16 + ln) * 32 + 16 + lg * 8]);
      acc[nt] = __builtin_amdgcn_wmma_f32_16x16x32_f16(false, pa.v, false, bv.v,
                                                       (short)0, acc[nt], false, false);
    }
  };

  stageKV(0, klo);
  for (int i = 0; i < nk - 1; ++i) {
    stageKV((i + 1) & 1, klo + ((i + 1) << 5));
    WAIT_ASYNC(8);        // previous batch complete, next still in flight
    __syncthreads();
    process(i & 1, klo + (i << 5));
    __syncthreads();
  }
  WAIT_ASYNC(0);
  __syncthreads();
  process((nk - 1) & 1, klo + ((nk - 1) << 5));

  float oscale = (tt == 1) ? tanhf(gamma[h]) : 1.f;
  float linv = 1.f / lsum;
#pragma unroll
  for (int r = 0; r < 8; ++r) {
    float li = __shfl(linv, r + 8 * lg, 32);
    int qr = qw + r + 8 * lg;
    _Float16* orow = Oc + (size_t)qr * (2 * NH * HD) + tt * (NH * HD) + h * HD;
#pragma unroll
    for (int nt = 0; nt < 8; ++nt)
      orow[nt * 16 + ln] = (_Float16)(acc[nt][r] * li * oscale);
  }
}

// ---------------------------------------------------------------------------
extern "C" void kernel_launch(void* const* d_in, const int* in_sizes, int n_in,
                              void* d_out, int out_size, void* d_ws, size_t ws_size,
                              hipStream_t stream) {
  const float* hs    = (const float*)d_in[0];
  const float* cosb  = (const float*)d_in[1];
  const float* sinb  = (const float*)d_in[2];
  const float* Wq    = (const float*)d_in[3];
  const float* bq    = (const float*)d_in[4];
  const float* Wk    = (const float*)d_in[5];
  const float* bk    = (const float*)d_in[6];
  const float* Wv    = (const float*)d_in[7];
  const float* bv    = (const float*)d_in[8];
  const float* Wo    = (const float*)d_in[9];
  const float* gamma = (const float*)d_in[10];
  float* out = (float*)d_out;

  char* ws = (char*)d_ws;
  size_t off = 0;
  auto alloc = [&](size_t bytes) -> void* {
    void* p = ws + off;
    off = (off + bytes + 255) & ~(size_t)255;
    return p;
  };
  _Float16* Xh  = (_Float16*)alloc((size_t)S_LEN * HID * 2);            // 8 MB
  _Float16* Wqt = (_Float16*)alloc((size_t)2 * NH * HD * HID * 2);      // 16 MB
  _Float16* Wkt = (_Float16*)alloc((size_t)NKV * HD * HID * 2);         // 2 MB
  _Float16* Wvt = (_Float16*)alloc((size_t)NKV * HD * HID * 2);         // 2 MB
  _Float16* Wot = (_Float16*)alloc((size_t)HID * 2 * NH * HD * 2);      // 16 MB
  _Float16* Qp  = (_Float16*)alloc((size_t)S_LEN * 2 * NH * HD * 2);    // 16 MB
  _Float16* Kp  = (_Float16*)alloc((size_t)S_LEN * NKV * HD * 2);       // 2 MB
  _Float16* Vp  = (_Float16*)alloc((size_t)S_LEN * NKV * HD * 2);       // 2 MB
  _Float16* Qh  = (_Float16*)alloc((size_t)2 * NH * S_LEN * HD * 2);    // 16 MB
  _Float16* Khb = (_Float16*)alloc((size_t)NKV * S_LEN * HD * 2);       // 2 MB
  _Float16* Vtb = (_Float16*)alloc((size_t)NKV * HD * S_LEN * 2);       // 2 MB
  _Float16* Oc  = Qp;  // alias: Qp is dead after rope_q, Oc written after

  // f16 conversions / weight transposes
  cvt_f16_kernel<<<(S_LEN * HID + 255) / 256, 256, 0, stream>>>(hs, Xh, S_LEN * HID);
  transpose_cvt_kernel<<<dim3(128, 64), dim3(32, 8), 0, stream>>>(Wq, Wqt, HID, 2 * NH * HD);
  transpose_cvt_kernel<<<dim3(16, 64),  dim3(32, 8), 0, stream>>>(Wk, Wkt, HID, NKV * HD);
  transpose_cvt_kernel<<<dim3(16, 64),  dim3(32, 8), 0, stream>>>(Wv, Wvt, HID, NKV * HD);
  transpose_cvt_kernel<<<dim3(64, 128), dim3(32, 8), 0, stream>>>(Wo, Wot, 2 * NH * HD, HID);

  // QKV projections (WMMA GEMM, f16 out)
  gemm_kernel<<<dim3(16, 32), 256, 0, stream>>>(Xh, Wqt, bq, Qp, S_LEN, 2 * NH * HD, HID, 1);
  gemm_kernel<<<dim3(16, 4),  256, 0, stream>>>(Xh, Wkt, bk, Kp, S_LEN, NKV * HD, HID, 1);
  gemm_kernel<<<dim3(16, 4),  256, 0, stream>>>(Xh, Wvt, bv, Vp, S_LEN, NKV * HD, HID, 1);

  // RoPE + attention-friendly layouts
  rope_q_kernel<<<dim3(S_LEN, 2 * NH), 128, 0, stream>>>(Qp, cosb, sinb, Qh);
  rope_k_kernel<<<dim3(S_LEN, NKV), 128, 0, stream>>>(Kp, cosb, sinb, Khb);
  transpose_f16_kernel<<<dim3(16, 64), dim3(32, 8), 0, stream>>>(Vp, Vtb, S_LEN, NKV * HD);

  // dual-mask flash attention (WMMA), writes concatenated O [S, 4096] f16
  attn_kernel<<<dim3(S_LEN / 64, NH, 2), 128, 0, stream>>>(Qh, Khb, Vtb, gamma, Oc);

  // output projection (WMMA GEMM, f32 out)
  gemm_kernel<<<dim3(16, 16), 256, 0, stream>>>(Oc, Wot, nullptr, out, S_LEN, HID, 2 * NH * HD, 0);
}
